// ASAP_43456479101194
// MI455X (gfx1250) — matrix operand order repaired and checked by reading, not compiled, not run on hardware
//
#include <hip/hip_runtime.h>
#include <math.h>

// ---------------- problem constants ----------------
#define Nn   4096
#define Ee   131072
#define Fin  128
#define Hh   64
#define Kk   3277          // ceil(0.8*N)
#define KP   3328          // K padded to multiple of 64 (wave tile = 16x64)
#define ETot (Ee + Nn)     // edges + self loops

typedef __bf16 bf16;
typedef __attribute__((ext_vector_type(16))) __bf16 v16bf;
typedef __attribute__((ext_vector_type(8)))  __bf16 v8bf;
typedef __attribute__((ext_vector_type(4)))  __bf16 v4bf;
typedef __attribute__((ext_vector_type(8)))  float  v8f;
typedef __attribute__((ext_vector_type(4)))  float  v4f;

// ---------------- small helpers ----------------
__device__ inline void edge_sd(int e, const int* __restrict__ src0,
                               const int* __restrict__ dst0, int& s, int& d) {
  if (e < Ee) { s = src0[e]; d = dst0[e]; }
  else        { int v = e - Ee; s = v; d = v; }   // self loop
}

__device__ inline void atomicMaxF(float* addr, float val) {
  unsigned int* ua = (unsigned int*)addr;
  unsigned int old = __float_as_uint(*addr);
  while (__uint_as_float(old) < val) {
    unsigned int assumed = old;
    old = atomicCAS(ua, assumed, __float_as_uint(val));
    if (old == assumed) break;
  }
}

// ---------------- generic elementwise (vectorized: n must be multiple of 4) ----------------
__global__ void k_fill4(float* __restrict__ p, float v, int n4) {
  v4f w = {v, v, v, v};
  v4f* p4 = (v4f*)p;
  for (int i = blockIdx.x * blockDim.x + threadIdx.x; i < n4; i += gridDim.x * blockDim.x)
    p4[i] = w;
}

__global__ void k_cvt_bf16_4(const float* __restrict__ in, bf16* __restrict__ out, int n4) {
  const v4f* i4 = (const v4f*)in;
  v4bf* o4 = (v4bf*)out;
  for (int i = blockIdx.x * blockDim.x + threadIdx.x; i < n4; i += gridDim.x * blockDim.x) {
    v4f v = i4[i];
    v4bf o;
    o[0] = (bf16)v[0]; o[1] = (bf16)v[1]; o[2] = (bf16)v[2]; o[3] = (bf16)v[3];
    o4[i] = o;
  }
}

__global__ void k_relu(const float* __restrict__ in, float* __restrict__ out, int n) {
  for (int i = blockIdx.x * blockDim.x + threadIdx.x; i < n; i += gridDim.x * blockDim.x)
    out[i] = fmaxf(in[i], 0.f);
}

// ---------------- WMMA bf16 GEMM: C[M,N] = A[M,K] * B[K,N] (+bias) ----------------
// Row-major A (lda) and B (ldb). Each wave computes a 16x64 output strip:
// the A fragment is reused by 4 WMMAs into 4 independent f32 accumulators.
// Requirements: M % 16 == 0, Ncol % 64 == 0, Kd % 32 == 0, lda/ldb % 16 == 0.
__global__ void k_wmma_gemm(const bf16* __restrict__ A, const bf16* __restrict__ B,
                            float* __restrict__ C, const float* __restrict__ bias,
                            int M, int Ncol, int Kd, int lda, int ldb, int ldc,
                            int useBias) {
  const int lane = threadIdx.x & 31;
  const int wave = threadIdx.x >> 5;
  const int tilesN = Ncol >> 6;                    // 64-wide strips
  const int tilesM = M >> 4;
  const int tile = blockIdx.x * (blockDim.x >> 5) + wave;
  if (tile >= tilesM * tilesN) return;
  const int tm = (tile / tilesN) << 4;
  const int tn = (tile % tilesN) << 6;
  const int g = lane >> 4;                         // half-wave group
  const int r = lane & 15;

  v8f acc0 = {}, acc1 = {}, acc2 = {}, acc3 = {};
  const bf16* arow = A + (size_t)(tm + r) * lda;
  for (int kk = 0; kk < Kd; kk += 32) {
    // A fragment: lane holds row (tm+r); K runs [kk+8g, +8) and [kk+16+8g, +8)
    v8bf alo = *(const v8bf*)(arow + kk + 8 * g);
    v8bf ahi = *(const v8bf*)(arow + kk + 16 + 8 * g);
    v16bf av = __builtin_shufflevector(alo, ahi,
                                       0,1,2,3,4,5,6,7,8,9,10,11,12,13,14,15);
    // B fragments: lane holds column (tn+16q+r); K runs [kk+8g, +8) and [kk+16+8g, +8)
    const bf16* bp = B + (size_t)kk * ldb + tn + r;
    v16bf bv0, bv1, bv2, bv3;
#pragma unroll
    for (int s = 0; s < 16; ++s) {
      const int k = (s < 8) ? (8 * g + s) : (8 + 8 * g + s);
      const bf16* p = bp + (size_t)k * ldb;
      bv0[s] = p[0]; bv1[s] = p[16]; bv2[s] = p[32]; bv3[s] = p[48];
    }

    if (kk + 32 < Kd)
      __builtin_prefetch(B + (size_t)(kk + 32 + lane) * ldb + tn, 0, 1);

    acc0 = __builtin_amdgcn_wmma_f32_16x16x32_bf16(false, av, false, bv0, (short)0, acc0, false, false);
    acc1 = __builtin_amdgcn_wmma_f32_16x16x32_bf16(false, av, false, bv1, (short)0, acc1, false, false);
    acc2 = __builtin_amdgcn_wmma_f32_16x16x32_bf16(false, av, false, bv2, (short)0, acc2, false, false);
    acc3 = __builtin_amdgcn_wmma_f32_16x16x32_bf16(false, av, false, bv3, (short)0, acc3, false, false);
  }
  // store: VGPR p -> row tm + p + 8g; col tn + 16q + r
#pragma unroll
  for (int q = 0; q < 4; ++q) {
    const v8f* accp = (q == 0) ? &acc0 : (q == 1) ? &acc1 : (q == 2) ? &acc2 : &acc3;
    v8f a = *accp;
    float bval = useBias ? bias[tn + 16 * q + r] : 0.f;
#pragma unroll
    for (int p = 0; p < 8; ++p)
      C[(size_t)(tm + p + 8 * g) * ldc + tn + 16 * q + r] = a[p] + bval;
  }
}

// ---------------- edge scatter kernels ----------------
__global__ void k_seg_edge_accum(const float* __restrict__ a, const float* __restrict__ b,
                                 const int* __restrict__ src0, const int* __restrict__ dst0,
                                 float* __restrict__ acc, int nE) {
  int t = blockIdx.x * blockDim.x + threadIdx.x;
  if (t >= nE * Hh) return;
  int e = t / Hh, h = t - e * Hh, s, d;
  edge_sd(e, src0, dst0, s, d);
  atomicAdd(&acc[(size_t)d * Hh + h], a[(size_t)s * Hh + h] - b[(size_t)d * Hh + h]);
}

__global__ void k_seg_max_feat(const float* __restrict__ x,
                               const int* __restrict__ src0, const int* __restrict__ dst0,
                               float* __restrict__ xq, int nE) {
  int t = blockIdx.x * blockDim.x + threadIdx.x;
  if (t >= nE * Hh) return;
  int e = t / Hh, h = t - e * Hh, s, d;
  edge_sd(e, src0, dst0, s, d);
  atomicMaxF(&xq[(size_t)d * Hh + h], x[(size_t)s * Hh + h]);
}

__global__ void k_att_proj(const float* __restrict__ xq, const float* __restrict__ x2,
                           const float* __restrict__ attw, float* __restrict__ p1,
                           float* __restrict__ p2) {
  int n = blockIdx.x * blockDim.x + threadIdx.x;
  if (n >= Nn) return;
  float s1 = 0.f, s2 = 0.f;
  for (int h = 0; h < Hh; ++h) {
    s1 += xq[(size_t)n * Hh + h] * attw[h];
    s2 += x2[(size_t)n * Hh + h] * attw[Hh + h];
  }
  p1[n] = s1; p2[n] = s2;
}

__global__ void k_raw_max(const float* __restrict__ p1, const float* __restrict__ p2,
                          const float* __restrict__ attb,
                          const int* __restrict__ src0, const int* __restrict__ dst0,
                          float* __restrict__ raw, float* __restrict__ m) {
  int e = blockIdx.x * blockDim.x + threadIdx.x;
  if (e >= ETot) return;
  int s, d; edge_sd(e, src0, dst0, s, d);
  float r = p1[d] + p2[s] + attb[0];
  r = (r > 0.f) ? r : 0.2f * r;
  raw[e] = r;
  atomicMaxF(&m[d], r);
}

__global__ void k_exp_den(const float* __restrict__ raw, const float* __restrict__ m,
                          const int* __restrict__ src0, const int* __restrict__ dst0,
                          float* __restrict__ score, float* __restrict__ den) {
  int e = blockIdx.x * blockDim.x + threadIdx.x;
  if (e >= ETot) return;
  int s, d; edge_sd(e, src0, dst0, s, d);
  float ex = expf(raw[e] - m[d]);
  score[e] = ex;
  atomicAdd(&den[d], ex);
}

__global__ void k_score_div(float* __restrict__ score, const float* __restrict__ den,
                            const int* __restrict__ src0, const int* __restrict__ dst0) {
  int e = blockIdx.x * blockDim.x + threadIdx.x;
  if (e >= ETot) return;
  int s, d; edge_sd(e, src0, dst0, s, d);
  score[e] = score[e] / (den[d] + 1e-16f);
}

__global__ void k_weighted_sum(const float* __restrict__ score, const float* __restrict__ x2,
                               const int* __restrict__ src0, const int* __restrict__ dst0,
                               float* __restrict__ xnew) {
  int t = blockIdx.x * blockDim.x + threadIdx.x;
  if (t >= ETot * Hh) return;
  int e = t / Hh, h = t - e * Hh, s, d;
  edge_sd(e, src0, dst0, s, d);
  atomicAdd(&xnew[(size_t)d * Hh + h], score[e] * x2[(size_t)s * Hh + h]);
}

__global__ void k_fit_proj(const float* __restrict__ xnew,
                           const float* __restrict__ w1, const float* __restrict__ b1,
                           const float* __restrict__ w2, const float* __restrict__ w3,
                           const float* __restrict__ b3,
                           float* __restrict__ af, float* __restrict__ bf_,
                           float* __restrict__ accf) {
  int n = blockIdx.x * blockDim.x + threadIdx.x;
  if (n >= Nn) return;
  float s1 = b1[0], s2 = 0.f, s3 = b3[0];
  for (int h = 0; h < Hh; ++h) {
    float v = xnew[(size_t)n * Hh + h];
    s1 += v * w1[h]; s2 += v * w2[h]; s3 += v * w3[h];
  }
  af[n] = s1; bf_[n] = s2; accf[n] = s3;
}

__global__ void k_fit_edge(const float* __restrict__ af, const float* __restrict__ bf_,
                           const int* __restrict__ src0, const int* __restrict__ dst0,
                           float* __restrict__ accf) {
  int e = blockIdx.x * blockDim.x + threadIdx.x;
  if (e >= ETot) return;
  int s, d; edge_sd(e, src0, dst0, s, d);
  atomicAdd(&accf[d], af[s] - bf_[d]);
}

__global__ void k_sigmoid(const float* __restrict__ in, float* __restrict__ out, int n) {
  int i = blockIdx.x * blockDim.x + threadIdx.x;
  if (i < n) out[i] = 1.f / (1.f + expf(-in[i]));
}

// ---------------- top-K via one-block bitonic sort (descending, stable-ish) ----------------
__global__ __launch_bounds__(1024) void k_topk(const float* __restrict__ fit,
                                               int* __restrict__ perm,
                                               int* __restrict__ colmap) {
  __shared__ float key[Nn];
  __shared__ int   idx[Nn];
  for (int i = threadIdx.x; i < Nn; i += 1024) { key[i] = fit[i]; idx[i] = i; }
  __syncthreads();
  for (int ks = 2; ks <= Nn; ks <<= 1) {
    for (int j = ks >> 1; j > 0; j >>= 1) {
      for (int t = threadIdx.x; t < Nn / 2; t += 1024) {
        int i  = ((t & ~(j - 1)) << 1) | (t & (j - 1));
        int p  = i | j;
        float ki = key[i], kp = key[p];
        int   ii = idx[i], ip = idx[p];
        bool lessI = (ki < kp) || (ki == kp && ii > ip);   // "less" in descending order
        bool desc  = ((i & ks) == 0);
        if (desc ? lessI : !lessI) {
          key[i] = kp; key[p] = ki; idx[i] = ip; idx[p] = ii;
        }
      }
      __syncthreads();
    }
  }
  for (int i = threadIdx.x; i < Nn; i += 1024) colmap[i] = -1;
  __syncthreads();
  for (int j = threadIdx.x; j < KP; j += 1024) perm[j] = (j < Kk) ? idx[j] : -1;
  __syncthreads();
  for (int j = threadIdx.x; j < Kk; j += 1024) colmap[idx[j]] = j;
}

__global__ void k_gather_xp(const float* __restrict__ xnew, const float* __restrict__ fit,
                            const int* __restrict__ perm, float* __restrict__ xp) {
  int t = blockIdx.x * blockDim.x + threadIdx.x;
  if (t >= KP * Hh) return;
  int j = t / Hh, h = t - j * Hh;
  float v = 0.f;
  if (j < Kk) { int n = perm[j]; v = xnew[(size_t)n * Hh + h] * fit[n]; }
  xp[(size_t)j * Hh + h] = v;
}

// Build A^T (dense), Ssel (N x KP) and Ssel^T (KP x N) from the edge list.
// AdT[d, s] = A[s, d]; Ssel[s, j] += score; SselT[j, s] += score  (j = colmap[d]).
__global__ void k_build_AS(const float* __restrict__ score,
                           const int* __restrict__ src0, const int* __restrict__ dst0,
                           const int* __restrict__ colmap,
                           float* __restrict__ AdT, float* __restrict__ Ssel,
                           float* __restrict__ SselT) {
  int e = blockIdx.x * blockDim.x + threadIdx.x;
  if (e >= ETot) return;
  int s, d; edge_sd(e, src0, dst0, s, d);
  atomicAdd(&AdT[(size_t)d * Nn + s], 1.0f);
  int j = colmap[d];
  if (j >= 0) {
    float sc = score[e];
    atomicAdd(&Ssel[(size_t)s * KP + j], sc);
    atomicAdd(&SselT[(size_t)j * Nn + s], sc);
  }
}

__global__ void k_mask_diag(float* __restrict__ A2T) {
  int i = blockIdx.x * blockDim.x + threadIdx.x;
  if (i < KP) A2T[(size_t)i * KP + i] = 0.f;
}

// deg[j] = sum_u A2[u, j] = row sum of A2^T (coalesced: one block per row)
__global__ void k_deg(const float* __restrict__ A2T, float* __restrict__ deg) {
  __shared__ float s[256];
  int j = blockIdx.x;
  float acc = 0.f;
  for (int u = threadIdx.x; u < KP; u += 256) acc += A2T[(size_t)j * KP + u];
  s[threadIdx.x] = acc;
  __syncthreads();
  for (int o = 128; o > 0; o >>= 1) {
    if ((int)threadIdx.x < o) s[threadIdx.x] += s[threadIdx.x + o];
    __syncthreads();
  }
  if (threadIdx.x == 0) deg[j] = s[0];
}

__global__ void k_dense_combine(const float* __restrict__ t, const float* __restrict__ deg,
                                const float* __restrict__ b, const float* __restrict__ c,
                                float* __restrict__ out) {
  int i = blockIdx.x * blockDim.x + threadIdx.x;
  if (i >= KP * Hh) return;
  int j = i / Hh;
  float v = (j < Kk) ? fmaxf(t[i] - deg[j] * b[i] + c[i], 0.f) : 0.f;
  out[i] = v;
}

__global__ void k_col_mean(const float* __restrict__ x, float* __restrict__ out,
                           int rows, float inv) {
  __shared__ float s[256];
  int h = blockIdx.x;
  float acc = 0.f;
  for (int r = threadIdx.x; r < rows; r += 256) acc += x[(size_t)r * Hh + h];
  s[threadIdx.x] = acc;
  __syncthreads();
  for (int o = 128; o > 0; o >>= 1) {
    if ((int)threadIdx.x < o) s[threadIdx.x] += s[threadIdx.x + o];
    __syncthreads();
  }
  if (threadIdx.x == 0) out[h] = s[0] * inv;
}

__global__ void k_head(const float* __restrict__ h, const float* __restrict__ w1,
                       const float* __restrict__ b1, const float* __restrict__ w2,
                       const float* __restrict__ b2, float* __restrict__ out) {
  __shared__ float t[Hh];
  int i = threadIdx.x;
  float acc = b1[i];
  for (int k = 0; k < 4 * Hh; ++k) acc += h[k] * w1[k * Hh + i];
  acc = fmaxf(acc, 0.f);
  t[i] = acc * w2[i];
  __syncthreads();
  for (int o = 32; o > 0; o >>= 1) {
    if (i < o) t[i] += t[i + o];
    __syncthreads();
  }
  if (i == 0) out[0] = t[0] + b2[0];
}

// ---------------- host launcher ----------------
extern "C" void kernel_launch(void* const* d_in, const int* in_sizes, int n_in,
                              void* d_out, int out_size, void* d_ws, size_t ws_size,
                              hipStream_t stream) {
  (void)in_sizes; (void)n_in; (void)out_size; (void)ws_size;
  const float* x      = (const float*)d_in[0];
  const int*   ei     = (const int*)d_in[1];
  const float* c1_w1  = (const float*)d_in[2];
  const float* c1_b1  = (const float*)d_in[3];
  const float* c1_w2  = (const float*)d_in[4];
  const float* c1_w3  = (const float*)d_in[5];
  const float* c1_b3  = (const float*)d_in[6];
  const float* cs_w1  = (const float*)d_in[7];
  const float* cs_b1  = (const float*)d_in[8];
  const float* cs_w2  = (const float*)d_in[9];
  const float* cs_w3  = (const float*)d_in[10];
  const float* cs_b3  = (const float*)d_in[11];
  const float* pl_lin_w = (const float*)d_in[12];
  const float* pl_lin_b = (const float*)d_in[13];
  const float* pl_att_w = (const float*)d_in[14];
  const float* pl_att_b = (const float*)d_in[15];
  const float* pg_w1  = (const float*)d_in[16];
  const float* pg_b1  = (const float*)d_in[17];
  const float* pg_w2  = (const float*)d_in[18];
  const float* pg_w3  = (const float*)d_in[19];
  const float* pg_b3  = (const float*)d_in[20];
  const float* l1_w   = (const float*)d_in[21];
  const float* l1_b   = (const float*)d_in[22];
  const float* l2_w   = (const float*)d_in[23];
  const float* l2_b   = (const float*)d_in[24];
  float* out = (float*)d_out;
  const int* src0 = ei;
  const int* dst0 = ei + Ee;

  // workspace allocator (256B aligned)
  char* wp = (char*)d_ws;
  auto alloc = [&](size_t bytes) -> void* {
    void* r = (void*)wp;
    wp += (bytes + 255) & ~(size_t)255;
    return r;
  };
  bf16* Xb   = (bf16*)alloc((size_t)Nn * Fin * 2);
  bf16* wAb  = (bf16*)alloc((size_t)Fin * Hh * 2);
  bf16* wBb  = (bf16*)alloc((size_t)Fin * Hh * 2);
  bf16* wCb  = (bf16*)alloc((size_t)Fin * Hh * 2);
  bf16* plwb = (bf16*)alloc((size_t)Hh * Hh * 2);
  float* bufA = (float*)alloc((size_t)Nn * Hh * 4);
  float* bufB = (float*)alloc((size_t)Nn * Hh * 4);
  float* bufC = (float*)alloc((size_t)Nn * Hh * 4);
  float* accv = (float*)alloc((size_t)Nn * Hh * 4);
  float* x1   = (float*)alloc((size_t)Nn * Hh * 4);
  float* x2   = (float*)alloc((size_t)Nn * Hh * 4);
  bf16* xb1   = (bf16*)alloc((size_t)Nn * Hh * 2);
  float* xq   = (float*)alloc((size_t)Nn * Hh * 4);
  bf16* xqb   = (bf16*)alloc((size_t)Nn * Hh * 2);
  float* xql  = (float*)alloc((size_t)Nn * Hh * 4);
  float* p1   = (float*)alloc((size_t)Nn * 4);
  float* p2   = (float*)alloc((size_t)Nn * 4);
  float* raw  = (float*)alloc((size_t)ETot * 4);
  float* mseg = (float*)alloc((size_t)Nn * 4);
  float* den  = (float*)alloc((size_t)Nn * 4);
  float* score = (float*)alloc((size_t)ETot * 4);
  float* xnew = (float*)alloc((size_t)Nn * Hh * 4);
  float* af   = (float*)alloc((size_t)Nn * 4);
  float* bf_  = (float*)alloc((size_t)Nn * 4);
  float* accf = (float*)alloc((size_t)Nn * 4);
  float* fit  = (float*)alloc((size_t)Nn * 4);
  int*  perm  = (int*)alloc((size_t)KP * 4);
  int*  colmap = (int*)alloc((size_t)Nn * 4);
  float* xp   = (float*)alloc((size_t)KP * Hh * 4);
  bf16* xpb   = (bf16*)alloc((size_t)KP * Hh * 2);
  float* AdT  = (float*)alloc((size_t)Nn * Nn * 4);
  bf16*  AdTb = (bf16*)alloc((size_t)Nn * Nn * 2);
  float* Ssel = (float*)alloc((size_t)Nn * KP * 4);
  bf16*  Sselb = (bf16*)alloc((size_t)Nn * KP * 2);
  float* SselT = (float*)alloc((size_t)KP * Nn * 4);
  bf16*  SselTb = (bf16*)alloc((size_t)KP * Nn * 2);
  float* MTf  = (float*)alloc((size_t)KP * Nn * 4);
  bf16*  MTb  = (bf16*)alloc((size_t)KP * Nn * 2);
  float* A2T  = (float*)alloc((size_t)KP * KP * 4);
  bf16*  A2Tb = (bf16*)alloc((size_t)KP * KP * 2);
  float* deg  = (float*)alloc((size_t)KP * 4);
  float* tmpD = (float*)alloc((size_t)KP * Hh * 4);
  bf16*  a3b  = (bf16*)alloc((size_t)KP * Hh * 2);
  float* x3   = (float*)alloc((size_t)KP * Hh * 4);
  float* x4   = (float*)alloc((size_t)KP * Hh * 4);
  bf16*  x3b  = (bf16*)alloc((size_t)KP * Hh * 2);
  float* hvec = (float*)alloc(4 * Hh * 4);

  auto gi = [](long long n, int b) { return (int)((n + b - 1) / b); };
  auto cvt = [&](const float* src, bf16* dst, long long n) {       // n % 4 == 0
    k_cvt_bf16_4<<<gi(n / 4, 256), 256, 0, stream>>>(src, dst, (int)(n / 4));
  };
  auto fill = [&](float* dst, float v, long long n) {              // n % 4 == 0
    k_fill4<<<gi(n / 4, 256), 256, 0, stream>>>(dst, v, (int)(n / 4));
  };
  auto gemm = [&](const bf16* A, const bf16* B, float* C, const float* bias,
                  int M, int Ncol, int Kd, int lda, int ldb, int ldc, int uB) {
    int tiles = (M / 16) * (Ncol / 64);
    k_wmma_gemm<<<(tiles + 7) / 8, 256, 0, stream>>>(A, B, C, bias, M, Ncol, Kd,
                                                     lda, ldb, ldc, uB);
  };

  // ===== LeConv 1 (edge, F_IN=128 -> H=64) =====
  cvt(x, Xb, (long long)Nn * Fin);
  cvt(c1_w1, wAb, Fin * Hh); cvt(c1_w2, wBb, Fin * Hh); cvt(c1_w3, wCb, Fin * Hh);
  gemm(Xb, wAb, bufA, c1_b1, Nn, Hh, Fin, Fin, Hh, Hh, 1);
  gemm(Xb, wBb, bufB, nullptr, Nn, Hh, Fin, Fin, Hh, Hh, 0);
  gemm(Xb, wCb, accv, c1_b3, Nn, Hh, Fin, Fin, Hh, Hh, 1);
  k_seg_edge_accum<<<gi((long long)Ee * Hh, 256), 256, 0, stream>>>(bufA, bufB, src0, dst0, accv, Ee);
  k_relu<<<gi((long long)Nn * Hh, 256), 256, 0, stream>>>(accv, x1, Nn * Hh);

  // ===== LeConv 2 (edge, H->H, cs layer 0) =====
  cvt(x1, xb1, (long long)Nn * Hh);
  cvt(cs_w1 + 0, wAb, Hh * Hh); cvt(cs_w2 + 0, wBb, Hh * Hh); cvt(cs_w3 + 0, wCb, Hh * Hh);
  gemm(xb1, wAb, bufA, cs_b1 + 0, Nn, Hh, Hh, Hh, Hh, Hh, 1);
  gemm(xb1, wBb, bufB, nullptr, Nn, Hh, Hh, Hh, Hh, Hh, 0);
  gemm(xb1, wCb, accv, cs_b3 + 0, Nn, Hh, Hh, Hh, Hh, Hh, 1);
  k_seg_edge_accum<<<gi((long long)Ee * Hh, 256), 256, 0, stream>>>(bufA, bufB, src0, dst0, accv, Ee);
  k_relu<<<gi((long long)Nn * Hh, 256), 256, 0, stream>>>(accv, x2, Nn * Hh);

  // ===== ASAP pool =====
  fill(xq, -3.0e38f, (long long)Nn * Hh);
  k_seg_max_feat<<<gi((long long)ETot * Hh, 256), 256, 0, stream>>>(x2, src0, dst0, xq, ETot);
  cvt(xq, xqb, (long long)Nn * Hh);
  cvt(pl_lin_w, plwb, Hh * Hh);
  gemm(xqb, plwb, xql, pl_lin_b, Nn, Hh, Hh, Hh, Hh, Hh, 1);
  k_att_proj<<<gi(Nn, 256), 256, 0, stream>>>(xql, x2, pl_att_w, p1, p2);
  fill(mseg, -3.0e38f, Nn);
  k_raw_max<<<gi(ETot, 256), 256, 0, stream>>>(p1, p2, pl_att_b, src0, dst0, raw, mseg);
  fill(den, 0.f, Nn);
  k_exp_den<<<gi(ETot, 256), 256, 0, stream>>>(raw, mseg, src0, dst0, score, den);
  k_score_div<<<gi(ETot, 256), 256, 0, stream>>>(score, den, src0, dst0);
  fill(xnew, 0.f, (long long)Nn * Hh);
  k_weighted_sum<<<gi((long long)ETot * Hh, 256), 256, 0, stream>>>(score, x2, src0, dst0, xnew);
  k_fit_proj<<<gi(Nn, 256), 256, 0, stream>>>(xnew, pg_w1, pg_b1, pg_w2, pg_w3, pg_b3, af, bf_, accf);
  k_fit_edge<<<gi(ETot, 256), 256, 0, stream>>>(af, bf_, src0, dst0, accf);
  k_sigmoid<<<gi(Nn, 256), 256, 0, stream>>>(accf, fit, Nn);
  k_topk<<<1, 1024, 0, stream>>>(fit, perm, colmap);
  k_gather_xp<<<gi((long long)KP * Hh, 256), 256, 0, stream>>>(xnew, fit, perm, xp);

  // ===== A2^T = (Ssel^T A^T) Ssel via transpose-free bf16 WMMA GEMMs =====
  fill(AdT, 0.f, (long long)Nn * Nn);
  fill(Ssel, 0.f, (long long)Nn * KP);
  fill(SselT, 0.f, (long long)KP * Nn);
  k_build_AS<<<gi(ETot, 256), 256, 0, stream>>>(score, src0, dst0, colmap, AdT, Ssel, SselT);
  cvt(AdT, AdTb, (long long)Nn * Nn);
  cvt(Ssel, Sselb, (long long)Nn * KP);
  cvt(SselT, SselTb, (long long)KP * Nn);
  gemm(SselTb, AdTb, MTf, nullptr, KP, Nn, Nn, Nn, Nn, Nn, 0);   // M^T = Ssel^T * A^T
  cvt(MTf, MTb, (long long)KP * Nn);
  gemm(MTb, Sselb, A2T, nullptr, KP, KP, Nn, Nn, KP, KP, 0);     // A2^T = M^T * Ssel
  k_mask_diag<<<gi(KP, 256), 256, 0, stream>>>(A2T);
  cvt(A2T, A2Tb, (long long)KP * KP);
  k_deg<<<KP, 256, 0, stream>>>(A2T, deg);

  // ===== dense LeConv layer 1 (cs layer 1) =====
  cvt(xp, xpb, (long long)KP * Hh);
  cvt(cs_w1 + Hh * Hh, wAb, Hh * Hh);
  cvt(cs_w2 + Hh * Hh, wBb, Hh * Hh);
  cvt(cs_w3 + Hh * Hh, wCb, Hh * Hh);
  gemm(xpb, wAb, bufA, cs_b1 + Hh, KP, Hh, Hh, Hh, Hh, Hh, 1);
  gemm(xpb, wBb, bufB, nullptr, KP, Hh, Hh, Hh, Hh, Hh, 0);
  gemm(xpb, wCb, bufC, cs_b3 + Hh, KP, Hh, Hh, Hh, Hh, Hh, 1);
  cvt(bufA, a3b, (long long)KP * Hh);
  gemm(A2Tb, a3b, tmpD, nullptr, KP, Hh, KP, KP, Hh, Hh, 0);     // A2^T * a
  k_dense_combine<<<gi((long long)KP * Hh, 256), 256, 0, stream>>>(tmpD, deg, bufB, bufC, x3);

  // ===== dense LeConv layer 2 (cs layer 2) =====
  cvt(x3, x3b, (long long)KP * Hh);
  cvt(cs_w1 + 2 * Hh * Hh, wAb, Hh * Hh);
  cvt(cs_w2 + 2 * Hh * Hh, wBb, Hh * Hh);
  cvt(cs_w3 + 2 * Hh * Hh, wCb, Hh * Hh);
  gemm(x3b, wAb, bufA, cs_b1 + 2 * Hh, KP, Hh, Hh, Hh, Hh, Hh, 1);
  gemm(x3b, wBb, bufB, nullptr, KP, Hh, Hh, Hh, Hh, Hh, 0);
  gemm(x3b, wCb, bufC, cs_b3 + 2 * Hh, KP, Hh, Hh, Hh, Hh, Hh, 1);
  cvt(bufA, a3b, (long long)KP * Hh);
  gemm(A2Tb, a3b, tmpD, nullptr, KP, Hh, KP, KP, Hh, Hh, 0);
  k_dense_combine<<<gi((long long)KP * Hh, 256), 256, 0, stream>>>(tmpD, deg, bufB, bufC, x4);

  // ===== readout =====
  k_col_mean<<<Hh, 256, 0, stream>>>(x1, hvec + 0 * Hh, Nn, 1.0f / Nn);
  k_col_mean<<<Hh, 256, 0, stream>>>(x2, hvec + 1 * Hh, Nn, 1.0f / Nn);
  k_col_mean<<<Hh, 256, 0, stream>>>(x3, hvec + 2 * Hh, KP, 1.0f / Kk);
  k_col_mean<<<Hh, 256, 0, stream>>>(x4, hvec + 3 * Hh, KP, 1.0f / Kk);
  k_head<<<1, Hh, 0, stream>>>(hvec, l1_w, l1_b, l2_w, l2_b, out);
}
